// Canglong_4844723109913
// MI455X (gfx1250) — matrix operand order repaired and checked by reading, not compiled, not run on hardware
//
#include <hip/hip_runtime.h>

typedef __attribute__((ext_vector_type(16))) _Float16 v16h;
typedef __attribute__((ext_vector_type(8)))  _Float16 v8h;
typedef __attribute__((ext_vector_type(8)))  float    v8f;
typedef __attribute__((ext_vector_type(4)))  float    v4f;

#define B_LON 30
#define NW    32
#define NTOK  144
#define DIMC  192
#define NHEAD 6
#define DHEAD 32
#define KPAD  160   // keys padded 144 -> 160 (5 chunks of K=32)

union V16U { v16h v; v8h h[2]; };

// A-matrix 16x32 f16 tile from row-major [*, ld] memory (ld in halves, mult of 8).
// lane<16: row=lane, K {0..7,16..23}; lane>=16: row=lane-16, K {8..15,24..31}.
__device__ __forceinline__ v16h load_a16x32(const _Float16* p, int ld) {
    unsigned lane = threadIdx.x & 31u;
    unsigned row  = lane & 15u;
    unsigned kb   = (lane >> 4) * 8u;
    const _Float16* q = p + (size_t)row * ld + kb;
    V16U r;
    r.h[0] = *(const v8h*)(q);
    r.h[1] = *(const v8h*)(q + 16);
    return r.v;
}

// B-matrix 32x16 f16 tile: B[k][n] = src[n*ld + k] (src row-major over N).
// lane<16: col=lane, K {0..7,16..23}; lane>=16: col=lane-16, K {8..15,24..31}.
__device__ __forceinline__ v16h load_b32x16(const _Float16* p, int ld) {
    unsigned lane = threadIdx.x & 31u;
    unsigned n    = lane & 15u;
    unsigned kb   = (lane >> 4) * 8u;
    const _Float16* q = p + (size_t)n * ld + kb;
    V16U r;
    r.h[0] = *(const v8h*)(q);
    r.h[1] = *(const v8h*)(q + 16);
    return r.v;
}

__device__ __forceinline__ v8f wmma_f16(v16h a, v16h b, v8f c) {
    return __builtin_amdgcn_wmma_f32_16x16x32_f16(false, a, false, b, (short)0, c, false, false);
}

// ---------- prep: transpose + f32->f16 weights (dst[c][r] = src[r][c]) ----------
__global__ void transpose_w_kernel(const float* __restrict__ src, _Float16* __restrict__ dst,
                                   int rows, int cols) {
    int idx = blockIdx.x * blockDim.x + threadIdx.x;
    if (idx >= rows * cols) return;
    int r = idx / cols, c = idx % cols;
    dst[(size_t)c * rows + r] = (_Float16)src[idx];
}

// ---------- prep: dense bias, TRANSPOSED layout ----------
// biasP[h][w][key k][query q] = bias_table[pos_index[q*144+k]][w][h]
__global__ void bias_prep_kernel(const float* __restrict__ bias_table,
                                 const int* __restrict__ pos_index,
                                 float* __restrict__ biasP) {
    int idx = blockIdx.x * blockDim.x + threadIdx.x;
    const int total = NHEAD * NW * NTOK * NTOK;
    if (idx >= total) return;
    int q = idx % NTOK;              // query (contiguous)
    int t = idx / NTOK;
    int k = t % NTOK; t /= NTOK;     // key
    int w = t % NW;
    int h = t / NW;
    int pi = pos_index[q * NTOK + k];
    biasP[idx] = bias_table[((size_t)pi * NW + w) * NHEAD + h];
}

// ---------- QKV projection: per-window GEMM [144,192] x [192,576] ----------
// Each wave owns 4 column tiles; weights stay in VGPRs, activations stream from LDS.
__global__ void __launch_bounds__(288)
qkv_kernel(const float* __restrict__ x, const _Float16* __restrict__ wqkvT,
           const float* __restrict__ b_qkv,
           _Float16* __restrict__ Qh, _Float16* __restrict__ Kh, _Float16* __restrict__ Vh) {
    __shared__ __align__(16) _Float16 xh[NTOK * DIMC];
    int bw = blockIdx.x;
    int b = bw / NW, w = bw % NW;
    const float* xp = x + (size_t)bw * NTOK * DIMC;
    for (int i = threadIdx.x; i < NTOK * DIMC; i += 288) xh[i] = (_Float16)xp[i];
    __syncthreads();

    int wave = threadIdx.x >> 5;
    unsigned lane = threadIdx.x & 31u;
    int col = lane & 15;
    int r0  = (lane >> 4) * 8;
    const float scale = 0.17677669529663687f;   // 32^-0.5

    for (int it = 0; it < 4; ++it) {
        int nt = wave * 4 + it;                  // this wave's column tile (0..35)

        // weight B-tiles for this column tile: loaded once, live across the M loop
        v16h Bt[6];
        #pragma unroll
        for (int kc = 0; kc < 6; ++kc)
            Bt[kc] = load_b32x16(wqkvT + (size_t)nt * 16 * DIMC + kc * 32, DIMC);

        // per-column-tile routing (constant within the tile: 16 cols never cross q/k/v or head)
        int j  = nt * 16 + col;                  // 0..575
        int t  = nt / 12;                        // 0:q 1:k 2:v
        int h  = (nt % 12) >> 1;
        int dd = ((nt & 1) << 4) + col;
        float bv = b_qkv[j];
        float s  = (t == 0) ? scale : 1.0f;
        _Float16* dst = (t == 0) ? Qh : (t == 1) ? Kh : Vh;
        size_t base = ((((size_t)b * NHEAD + h) * NW + w) * NTOK) * DHEAD + dd;

        for (int m = 0; m < 9; ++m) {            // all 9 row tiles
            v16h A[6];
            #pragma unroll
            for (int kc = 0; kc < 6; ++kc)
                A[kc] = load_a16x32(xh + (size_t)m * 16 * DIMC + kc * 32, DIMC);
            v8f acc = {};
            #pragma unroll
            for (int kc = 0; kc < 6; ++kc) acc = wmma_f16(A[kc], Bt[kc], acc);

            #pragma unroll
            for (int i = 0; i < 8; ++i)
                dst[base + (size_t)(m * 16 + r0 + i) * DHEAD] = (_Float16)((acc[i] + bv) * s);
        }
    }
}

// ---------- attention per (b,h,w): S = QK^T + bias + mask, softmax, O = P V ----------
__global__ void __launch_bounds__(288)
attn_kernel(const _Float16* __restrict__ Qh, const _Float16* __restrict__ Kh,
            const _Float16* __restrict__ Vh, const float* __restrict__ biasP,
            const float* __restrict__ mask, _Float16* __restrict__ Oh) {
    __shared__ __align__(16) _Float16 Vt[DHEAD * KPAD];        // V transposed, keys padded
    __shared__ __align__(16) _Float16 Ps[9 * 16 * KPAD];       // per-wave P blocks

    int id = blockIdx.x;
    int w = id % NW;
    int h = (id / NW) % NHEAD;
    int b = id / (NW * NHEAD);
    size_t qbase = ((((size_t)b * NHEAD + h) * NW + w) * NTOK) * DHEAD;

    const _Float16* Vsrc = Vh + qbase;
    for (int i = threadIdx.x; i < DHEAD * KPAD; i += 288) {
        int dd = i / KPAD, n = i % KPAD;
        Vt[i] = (n < NTOK) ? Vsrc[(size_t)n * DHEAD + dd] : (_Float16)0.f;
    }
    __syncthreads();

    int wave = threadIdx.x >> 5;
    int m0 = wave * 16;
    unsigned lane = threadIdx.x & 31u;
    int col = lane & 15;
    int r0  = (lane >> 4) * 8;

    // S = Q K^T : one WMMA per 16x16 tile (K = d = 32); B tiles in groups of 3
    v16h A = load_a16x32(Qh + qbase + (size_t)m0 * DHEAD, DHEAD);
    v8f S[9];
    #pragma unroll
    for (int g = 0; g < 3; ++g) {
        v16h B[3];
        #pragma unroll
        for (int u = 0; u < 3; ++u)
            B[u] = load_b32x16(Kh + qbase + (size_t)(g * 3 + u) * 16 * DHEAD, DHEAD);
        #pragma unroll
        for (int u = 0; u < 3; ++u) {
            v8f z = {};
            S[g * 3 + u] = wmma_f16(A, B[u], z);
        }
    }

    // bias (precomputed transposed [key][query]) + mask (symmetric -> read key-major):
    // per tile, each lane reads 8 contiguous queries -> b128 loads
    const float* bp = biasP + (((size_t)h * NW + w) * NTOK) * NTOK;
    const float* mp = mask  + (((size_t)b * NW + w) * NTOK) * NTOK;
    #pragma unroll
    for (int t = 0; t < 9; ++t) {
        int n = t * 16 + col;            // key index for this lane
        const v4f* b4 = (const v4f*)(bp + (size_t)n * NTOK + m0 + r0);
        const v4f* m4 = (const v4f*)(mp + (size_t)n * NTOK + m0 + r0);
        v4f b0 = b4[0], b1 = b4[1];
        v4f q0 = m4[0], q1 = m4[1];
        #pragma unroll
        for (int i = 0; i < 4; ++i) {
            S[t][i]     += b0[i] + q0[i];
            S[t][i + 4] += b1[i] + q1[i];
        }
    }

    // row softmax: row m lives in one 16-lane half; reduce via xor shuffles 1,2,4,8
    _Float16* Pw = Ps + (size_t)wave * 16 * KPAD;
    #pragma unroll
    for (int i = 0; i < 8; ++i) {
        float rmax = -3.0e38f;
        #pragma unroll
        for (int t = 0; t < 9; ++t) rmax = fmaxf(rmax, S[t][i]);
        #pragma unroll
        for (int off = 1; off < 16; off <<= 1) rmax = fmaxf(rmax, __shfl_xor(rmax, off, 32));
        float e[9], rsum = 0.f;
        #pragma unroll
        for (int t = 0; t < 9; ++t) { e[t] = __expf(S[t][i] - rmax); rsum += e[t]; }
        #pragma unroll
        for (int off = 1; off < 16; off <<= 1) rsum += __shfl_xor(rsum, off, 32);
        float rinv = 1.0f / rsum;
        int row = r0 + i;
        #pragma unroll
        for (int t = 0; t < 9; ++t)
            Pw[(size_t)row * KPAD + t * 16 + col] = (_Float16)(e[t] * rinv);
        Pw[(size_t)row * KPAD + NTOK + col] = (_Float16)0.f;   // zero the key padding
    }

    // O = P V : preload all 5 A chunks (shared by both column tiles), then per tile 5 B + 5 WMMAs
    v16h Ap[5];
    #pragma unroll
    for (int kc = 0; kc < 5; ++kc) Ap[kc] = load_a16x32(Pw + kc * 32, KPAD);

    size_t obase = (((size_t)b * NW + w) * NTOK) * DIMC + (size_t)h * DHEAD;
    #pragma unroll
    for (int nt = 0; nt < 2; ++nt) {
        v16h Bv[5];
        #pragma unroll
        for (int kc = 0; kc < 5; ++kc)
            Bv[kc] = load_b32x16(Vt + (size_t)nt * 16 * KPAD + kc * 32, KPAD);
        v8f acc = {};
        #pragma unroll
        for (int kc = 0; kc < 5; ++kc) acc = wmma_f16(Ap[kc], Bv[kc], acc);
        #pragma unroll
        for (int i = 0; i < 8; ++i)
            Oh[obase + (size_t)(m0 + r0 + i) * DIMC + nt * 16 + col] = (_Float16)acc[i];
    }
}

// ---------- output projection: per-window [144,192] x [192,192] + bias -> f32 ----------
// Each wave owns column tiles (strided); weights in VGPRs, activations from LDS.
__global__ void __launch_bounds__(288)
proj_kernel(const _Float16* __restrict__ Oh, const _Float16* __restrict__ wprojT,
            const float* __restrict__ b_proj, float* __restrict__ out) {
    __shared__ __align__(16) _Float16 Os[NTOK * DIMC];
    int bw = blockIdx.x;
    const _Float16* op = Oh + (size_t)bw * NTOK * DIMC;
    for (int i = threadIdx.x; i < NTOK * DIMC; i += 288) Os[i] = op[i];
    __syncthreads();

    int wave = threadIdx.x >> 5;
    unsigned lane = threadIdx.x & 31u;
    int col = lane & 15;
    int r0  = (lane >> 4) * 8;

    float* outp = out + (size_t)bw * NTOK * DIMC;
    for (int nt = wave; nt < 12; nt += 9) {
        v16h Bt[6];
        #pragma unroll
        for (int kc = 0; kc < 6; ++kc)
            Bt[kc] = load_b32x16(wprojT + (size_t)nt * 16 * DIMC + kc * 32, DIMC);
        float bb = b_proj[nt * 16 + col];

        for (int m = 0; m < 9; ++m) {
            v16h A[6];
            #pragma unroll
            for (int kc = 0; kc < 6; ++kc)
                A[kc] = load_a16x32(Os + (size_t)m * 16 * DIMC + kc * 32, DIMC);
            v8f acc = {};
            #pragma unroll
            for (int kc = 0; kc < 6; ++kc) acc = wmma_f16(A[kc], Bt[kc], acc);

            #pragma unroll
            for (int i = 0; i < 8; ++i)
                outp[(size_t)(m * 16 + r0 + i) * DIMC + nt * 16 + col] = acc[i] + bb;
        }
    }
}

extern "C" void kernel_launch(void* const* d_in, const int* in_sizes, int n_in,
                              void* d_out, int out_size, void* d_ws, size_t ws_size,
                              hipStream_t stream) {
    const float* x          = (const float*)d_in[0];
    const float* mask       = (const float*)d_in[1];
    const float* w_qkv      = (const float*)d_in[2];
    const float* b_qkv      = (const float*)d_in[3];
    const float* bias_table = (const float*)d_in[4];
    const float* w_proj     = (const float*)d_in[5];
    const float* b_proj     = (const float*)d_in[6];
    const int*   pos_index  = (const int*)d_in[7];
    float* out = (float*)d_out;

    char* ws = (char*)d_ws;
    const size_t SZ_WQKVT  = (size_t)576 * DIMC * 2;                              // 221184
    const size_t SZ_WPROJT = (size_t)DIMC * DIMC * 2;                             // 73728
    const size_t SZ_BIASP  = (size_t)NHEAD * NW * NTOK * NTOK * 4;                // 15925248
    const size_t SZ_QKV1   = (size_t)B_LON * NHEAD * NW * NTOK * DHEAD * 2;       // 53084160
    const size_t SZ_OH     = (size_t)B_LON * NW * NTOK * DIMC * 2;                // 53084160

    size_t off = 0;
    _Float16* wqkvT  = (_Float16*)(ws + off); off += SZ_WQKVT;
    _Float16* wprojT = (_Float16*)(ws + off); off += SZ_WPROJT;
    float*    biasP  = (float*)   (ws + off); off += SZ_BIASP;
    _Float16* Qh     = (_Float16*)(ws + off); off += SZ_QKV1;
    _Float16* Kh     = (_Float16*)(ws + off); off += SZ_QKV1;
    _Float16* Vh     = (_Float16*)(ws + off); off += SZ_QKV1;
    _Float16* Oh     = (_Float16*)(ws + off); off += SZ_OH;

    // weight transposes
    {
        int n1 = DIMC * 576;
        transpose_w_kernel<<<(n1 + 255) / 256, 256, 0, stream>>>(w_qkv, wqkvT, DIMC, 576);
        int n2 = DIMC * DIMC;
        transpose_w_kernel<<<(n2 + 255) / 256, 256, 0, stream>>>(w_proj, wprojT, DIMC, DIMC);
    }
    // dense position bias (transposed [key][query] layout)
    {
        int n = NHEAD * NW * NTOK * NTOK;
        bias_prep_kernel<<<(n + 255) / 256, 256, 0, stream>>>(bias_table, pos_index, biasP);
    }
    // QKV projection: one block per (b, w)
    qkv_kernel<<<B_LON * NW, 288, 0, stream>>>(x, wqkvT, b_qkv, Qh, Kh, Vh);
    // attention: one block per (b, h, w)
    attn_kernel<<<B_LON * NHEAD * NW, 288, 0, stream>>>(Qh, Kh, Vh, biasP, mask, Oh);
    // output projection: one block per (b, w)
    proj_kernel<<<B_LON * NW, 288, 0, stream>>>(Oh, wprojT, b_proj, out);
}
